// T5Attention_60198261621430
// MI455X (gfx1250) — compile-verified, matrix operand order checked by reading, not run on hardware
//
#include <hip/hip_runtime.h>
#include <hip/hip_bf16.h>
#include <hip/hip_fp16.h>

typedef __attribute__((ext_vector_type(16))) _Float16 v16h;
typedef __attribute__((ext_vector_type(8)))  _Float16 v8h;
typedef __attribute__((ext_vector_type(8)))  float    v8f;

#define D_MODEL   1024
#define NUM_HEADS 16
#define D_HEAD    64
#define MAX_DIST  128
#define SEQ       2048

#define LDS_STRIDE 40          // 32 halves + 8 pad -> conflict-free b128 reads
#define TILE_H     (128 * LDS_STRIDE)

// ---- fragment loaders -------------------------------------------------------
// 16x32 f16 A-fragment (or B-fragment of an n-major B^T) from row-major global.
// ISA 7.12.2: lane L holds row M=L%16; elems 0..7 = K base..base+7,
// elems 8..15 = K base+16..base+23, base = kbase + (L>=16 ? 8 : 0).
__device__ __forceinline__ v16h load_frag(const _Float16* __restrict__ base,
                                          int row0, int ld, int kbase, int lane) {
  const int m    = lane & 15;
  const int koff = kbase + ((lane & 16) >> 1);
  const _Float16* p = base + (size_t)(row0 + m) * (size_t)ld + koff;
  v8h lo = *(const v8h*)(p);
  v8h hi = *(const v8h*)(p + 16);
  v16h r;
#pragma unroll
  for (int i = 0; i < 8; ++i) { r[i] = lo[i]; r[i + 8] = hi[i]; }
  return r;
}

// Same fragment out of an LDS tile (rows padded to LDS_STRIDE halves).
__device__ __forceinline__ v16h lds_frag(const _Float16* t, int row0, int lane) {
  const int m    = lane & 15;
  const int koff = (lane & 16) >> 1;
  const _Float16* p = t + (row0 + m) * LDS_STRIDE + koff;
  v8h lo = *(const v8h*)(p);
  v8h hi = *(const v8h*)(p + 16);
  v16h r;
#pragma unroll
  for (int i = 0; i < 8; ++i) { r[i] = lo[i]; r[i + 8] = hi[i]; }
  return r;
}

__device__ __forceinline__ v8f wmma_f16(v16h a, v16h b, v8f c) {
  return __builtin_amdgcn_wmma_f32_16x16x32_f16(false, a, false, b, (short)0, c,
                                                false, false);
}

__device__ __forceinline__ void async_b128(unsigned lds, const _Float16* gp) {
  asm volatile("global_load_async_to_lds_b128 %0, %1, off"
               :: "v"(lds), "v"(gp) : "memory");
}

__device__ __forceinline__ void wait_async() {
  asm volatile("s_wait_asynccnt 0x0" ::: "memory");
}

// ---------------------------------------------------------------- converts --
__global__ void k_f32_to_f16(const float* __restrict__ in,
                             _Float16* __restrict__ out, int n) {
  int i = blockIdx.x * blockDim.x + threadIdx.x;
  int stride = gridDim.x * blockDim.x;
  for (; i < n; i += stride) out[i] = (_Float16)in[i];
}

// in[rows][cols] f32 -> out[cols][rows] f16
__global__ void k_transpose_f16(const float* __restrict__ in,
                                _Float16* __restrict__ out,
                                int rows, int cols) {
  __shared__ float tile[32][33];
  const int cb = blockIdx.x * 32, rb = blockIdx.y * 32;
  const int tx = threadIdx.x, ty = threadIdx.y;
  tile[ty][tx] = in[(size_t)(rb + ty) * cols + (cb + tx)];
  __syncthreads();
  out[(size_t)(cb + ty) * rows + (rb + tx)] = (_Float16)tile[tx][ty];
}

// ---- shared 128x128-block GEMM mainloop (async double-buffered LDS) --------
// Block: 256 threads / 8 waves; wave tile 64(M)x32(N) -> acc[4][2].
// K-loop manually unrolled by 2 so both buffers are compile-time offsets.
__device__ __forceinline__ void gemm_step(const _Float16* At, const _Float16* Bt,
                                          int wm, int wn, int lane,
                                          v8f acc[4][2]) {
  v16h b0 = lds_frag(Bt, wn * 32,      lane);
  v16h b1 = lds_frag(Bt, wn * 32 + 16, lane);
#pragma unroll
  for (int mi = 0; mi < 4; ++mi) {
    v16h a = lds_frag(At, wm * 64 + mi * 16, lane);
    acc[mi][0] = wmma_f16(a, b0, acc[mi][0]);
    acc[mi][1] = wmma_f16(a, b1, acc[mi][1]);
  }
}

__device__ __forceinline__ void gemm_mainloop(const _Float16* __restrict__ Ag,
                                              const _Float16* __restrict__ Bg,
                                              int mb, int nb,
                                              _Float16* sA, _Float16* sB,
                                              v8f acc[4][2]) {
  const int tid  = threadIdx.x;
  const int lane = tid & 31;
  const int wave = tid >> 5;
  const int wm = wave & 1;        // 2 M sub-tiles of 64
  const int wn = wave >> 1;       // 4 N sub-tiles of 32

  // Per-thread DMA coordinates: each thread copies two 16B chunks per tile
  // (rows r and r+64, 16B segment s of the 64B row-slice).
  const int r0 = tid >> 2, seg = (tid & 3) * 8;      // seg in halves
  const _Float16* gA0 = Ag + (size_t)(mb + r0)      * D_MODEL + seg;
  const _Float16* gA1 = Ag + (size_t)(mb + r0 + 64) * D_MODEL + seg;
  const _Float16* gB0 = Bg + (size_t)(nb + r0)      * D_MODEL + seg;
  const _Float16* gB1 = Bg + (size_t)(nb + r0 + 64) * D_MODEL + seg;
  const unsigned lA0a = (unsigned)(size_t)(sA + r0 * LDS_STRIDE + seg);
  const unsigned lA1a = (unsigned)(size_t)(sA + (r0 + 64) * LDS_STRIDE + seg);
  const unsigned lB0a = (unsigned)(size_t)(sB + r0 * LDS_STRIDE + seg);
  const unsigned lB1a = (unsigned)(size_t)(sB + (r0 + 64) * LDS_STRIDE + seg);
  const unsigned bufB = (unsigned)(TILE_H * sizeof(_Float16));  // buffer-1 delta

  // prefetch k=0 into buffer 0
  async_b128(lA0a, gA0); async_b128(lA1a, gA1);
  async_b128(lB0a, gB0); async_b128(lB1a, gB1);

  for (int k = 0; k < D_MODEL; k += 64) {
    // ---- half 1: compute buffer 0, refill buffer 1 with k+32 (always valid)
    wait_async();
    __syncthreads();
    async_b128(lA0a + bufB, gA0 + k + 32); async_b128(lA1a + bufB, gA1 + k + 32);
    async_b128(lB0a + bufB, gB0 + k + 32); async_b128(lB1a + bufB, gB1 + k + 32);
    gemm_step(sA, sB, wm, wn, lane, acc);

    // ---- half 2: compute buffer 1, refill buffer 0 with k+64
    wait_async();
    __syncthreads();
    if (k + 64 < D_MODEL) {
      async_b128(lA0a, gA0 + k + 64); async_b128(lA1a, gA1 + k + 64);
      async_b128(lB0a, gB0 + k + 64); async_b128(lB1a, gB1 + k + 64);
    }
    gemm_step(sA + TILE_H, sB + TILE_H, wm, wn, lane, acc);
  }
}

// ------------------------------------------------------------- QKV GEMM -----
__global__ void k_qkv_gemm(const _Float16* __restrict__ xh,
                           const _Float16* __restrict__ wT,   // [3072][1024]
                           _Float16* __restrict__ Qh,         // [16][2048][64]
                           _Float16* __restrict__ Kh,         // [16][2048][64]
                           _Float16* __restrict__ Vt) {       // [16][64][2048]
  __shared__ _Float16 sA[2 * TILE_H];
  __shared__ _Float16 sB[2 * TILE_H];
  const int lane = threadIdx.x & 31;
  const int wave = threadIdx.x >> 5;
  const int wm = wave & 1, wn = wave >> 1;
  const int mb = blockIdx.x * 128;
  const int nb = blockIdx.y * 128;

  v8f acc[4][2] = {};
  gemm_mainloop(xh, wT, mb, nb, sA, sB, acc);

  const int mofs  = (lane & 16) >> 1;
  const int nlane = lane & 15;
#pragma unroll
  for (int mi = 0; mi < 4; ++mi) {
#pragma unroll
    for (int ni = 0; ni < 2; ++ni) {
      const int ncol = nb + wn * 32 + ni * 16 + nlane;
      const int sec  = ncol >> 10;          // 0=Q 1=K 2=V
      const int cc   = ncol & 1023;
      const int h = cc >> 6, d = cc & 63;
#pragma unroll
      for (int r = 0; r < 8; ++r) {
        const int m = mb + wm * 64 + mi * 16 + r + mofs;
        const float v = acc[mi][ni][r];
        if (sec == 0)
          Qh[((size_t)h * SEQ + m) * D_HEAD + d] = (_Float16)(v * 0.125f);
        else if (sec == 1)
          Kh[((size_t)h * SEQ + m) * D_HEAD + d] = (_Float16)v;
        else
          Vt[((size_t)h * D_HEAD + d) * SEQ + m] = (_Float16)v;
      }
    }
  }
}

// ------------------------------------------------------------ out GEMM ------
__global__ void k_out_gemm(const _Float16* __restrict__ attnh,
                           const _Float16* __restrict__ woT,  // [1024][1024]
                           float* __restrict__ out) {         // [2048][1024]
  __shared__ _Float16 sA[2 * TILE_H];
  __shared__ _Float16 sB[2 * TILE_H];
  const int lane = threadIdx.x & 31;
  const int wave = threadIdx.x >> 5;
  const int wm = wave & 1, wn = wave >> 1;
  const int mb = blockIdx.x * 128;
  const int nb = blockIdx.y * 128;

  v8f acc[4][2] = {};
  gemm_mainloop(attnh, woT, mb, nb, sA, sB, acc);

  const int mofs  = (lane & 16) >> 1;
  const int nlane = lane & 15;
#pragma unroll
  for (int mi = 0; mi < 4; ++mi)
#pragma unroll
    for (int ni = 0; ni < 2; ++ni)
#pragma unroll
      for (int r = 0; r < 8; ++r) {
        const int m = mb + wm * 64 + mi * 16 + r + mofs;
        const int n = nb + wn * 32 + ni * 16 + nlane;
        out[(size_t)m * D_MODEL + n] = acc[mi][ni][r];
      }
}

// ------------------------------------------------------------- attention ----
// One block = one head x 32 query rows; 512 threads / 16 waves.
// Full 32x2048 f32 score panel in LDS (256KB of the 320KB WGP pool).
__global__ void k_attention(const _Float16* __restrict__ Qh,
                            const _Float16* __restrict__ Kh,
                            const _Float16* __restrict__ Vt,
                            const float* __restrict__ rel_bias,  // [128][16]
                            _Float16* __restrict__ attnh) {      // [2048][1024]
  extern __shared__ char smem[];
  float* sS    = (float*)smem;            // [32][2048]
  float* sBias = sS + 32 * SEQ;           // [128]
  float* sO    = sBias + MAX_DIST;        // [32][64]
  float* sInv  = sO + 32 * D_HEAD;        // [32]

  const int h    = blockIdx.y;
  const int qb   = blockIdx.x * 32;
  const int tid  = threadIdx.x;
  const int lane = tid & 31;
  const int wave = tid >> 5;              // 0..15
  const int mt   = wave & 1;              // which 16-row query sub-tile
  const int wkb  = wave >> 1;             // 0..7 key-tile stripe

  const _Float16* Q = Qh + (size_t)h * SEQ * D_HEAD;
  const _Float16* K = Kh + (size_t)h * SEQ * D_HEAD;
  const _Float16* V = Vt + (size_t)h * D_HEAD * SEQ;

  if (tid < MAX_DIST) sBias[tid] = rel_bias[tid * NUM_HEADS + h];
  for (int i = tid; i < 32 * D_HEAD; i += blockDim.x) sO[i] = 0.f;

  const int mofs  = (lane & 16) >> 1;
  const int nlane = lane & 15;

  // ---- phase 1: S = (Q/8) K^T + bias ----
  v16h a0 = load_frag(Q, qb + mt * 16, D_HEAD, 0,  lane);
  v16h a1 = load_frag(Q, qb + mt * 16, D_HEAD, 32, lane);
  __syncthreads();                        // sBias ready
  for (int kb = wkb; kb < SEQ / 16; kb += 8) {
    v16h b0 = load_frag(K, kb * 16, D_HEAD, 0,  lane);
    v16h b1 = load_frag(K, kb * 16, D_HEAD, 32, lane);
    v8f c = {};
    c = wmma_f16(a0, b0, c);
    c = wmma_f16(a1, b1, c);
    const int kpos = kb * 16 + nlane;
#pragma unroll
    for (int r = 0; r < 8; ++r) {
      const int m = mt * 16 + r + mofs;
      int dist = kpos - (qb + m);
      if (dist < 0) dist = -dist;
      if (dist > MAX_DIST - 1) dist = MAX_DIST - 1;
      sS[m * SEQ + kpos] = c[r] + sBias[dist];
    }
  }
  __syncthreads();

  // ---- phase 2: softmax (16 threads per row, 32 rows) ----
  {
    const int row = tid >> 4;
    const int l16 = tid & 15;
    float* Srow = sS + row * SEQ;
    float mx = -1e30f;
    for (int j = l16; j < SEQ; j += 16) mx = fmaxf(mx, Srow[j]);
#pragma unroll
    for (int o = 8; o > 0; o >>= 1) mx = fmaxf(mx, __shfl_xor(mx, o, 16));
    float sum = 0.f;
    for (int j = l16; j < SEQ; j += 16) {
      const float p = __expf(Srow[j] - mx);
      Srow[j] = p;
      sum += p;
    }
#pragma unroll
    for (int o = 8; o > 0; o >>= 1) sum += __shfl_xor(sum, o, 16);
    if (l16 == 0) sInv[row] = 1.f / sum;
  }
  __syncthreads();

  // ---- phase 3: O = P @ V ----
  v8f oacc[4] = {};
  const int m_a  = mt * 16 + (lane & 15);
  const int kofs = (lane & 16) >> 1;
  for (int kb = wkb; kb < SEQ / 32; kb += 8) {
    v16h p;
#pragma unroll
    for (int e = 0; e < 16; ++e) {
      const int kk = kb * 32 + kofs + (e < 8 ? e : e + 8);
      p[e] = (_Float16)sS[m_a * SEQ + kk];
    }
#pragma unroll
    for (int dc = 0; dc < 4; ++dc) {
      v16h b = load_frag(V, dc * 16, SEQ, kb * 32, lane);   // Vt contiguous
      oacc[dc] = wmma_f16(p, b, oacc[dc]);
    }
  }
#pragma unroll
  for (int dc = 0; dc < 4; ++dc)
#pragma unroll
    for (int r = 0; r < 8; ++r)
      atomicAdd(&sO[(mt * 16 + r + mofs) * D_HEAD + dc * 16 + nlane],
                oacc[dc][r]);
  __syncthreads();
  for (int i = tid; i < 32 * D_HEAD; i += blockDim.x) {
    const int m = i >> 6, d = i & 63;
    attnh[(size_t)(qb + m) * D_MODEL + h * D_HEAD + d] =
        (_Float16)(sO[i] * sInv[m]);
  }
}

// ---------------------------------------------------------------------------
extern "C" void kernel_launch(void* const* d_in, const int* in_sizes, int n_in,
                              void* d_out, int out_size, void* d_ws, size_t ws_size,
                              hipStream_t stream) {
  const float* x     = (const float*)d_in[0];   // [2048][1024]
  const float* w_qkv = (const float*)d_in[1];   // [1024][3072]
  const float* w_out = (const float*)d_in[2];   // [1024][1024]
  const float* rel_b = (const float*)d_in[3];   // [128][16]
  float* out = (float*)d_out;                   // [1][2048][1024] f32

  char* ws = (char*)d_ws;
  size_t o = 0;
  _Float16* xh    = (_Float16*)(ws + o); o += (size_t)SEQ * D_MODEL * 2;
  _Float16* wqkvT = (_Float16*)(ws + o); o += (size_t)3 * D_MODEL * D_MODEL * 2;
  _Float16* woT   = (_Float16*)(ws + o); o += (size_t)D_MODEL * D_MODEL * 2;
  _Float16* Qh    = (_Float16*)(ws + o); o += (size_t)NUM_HEADS * SEQ * D_HEAD * 2;
  _Float16* Kh    = (_Float16*)(ws + o); o += (size_t)NUM_HEADS * SEQ * D_HEAD * 2;
  _Float16* Vt    = (_Float16*)(ws + o); o += (size_t)NUM_HEADS * SEQ * D_HEAD * 2;
  _Float16* attnh = (_Float16*)(ws + o); o += (size_t)SEQ * D_MODEL * 2;
  (void)ws_size; (void)in_sizes; (void)n_in; (void)out_size;

  k_f32_to_f16<<<512, 256, 0, stream>>>(x, xh, SEQ * D_MODEL);

  dim3 tb(32, 32);
  k_transpose_f16<<<dim3(3 * D_MODEL / 32, D_MODEL / 32), tb, 0, stream>>>(
      w_qkv, wqkvT, D_MODEL, 3 * D_MODEL);
  k_transpose_f16<<<dim3(D_MODEL / 32, D_MODEL / 32), tb, 0, stream>>>(
      w_out, woT, D_MODEL, D_MODEL);

  k_qkv_gemm<<<dim3(SEQ / 128, 3 * D_MODEL / 128), 256, 0, stream>>>(
      xh, wqkvT, Qh, Kh, Vt);

  const size_t attn_lds =
      (size_t)(32 * SEQ + MAX_DIST + 32 * D_HEAD + 32) * sizeof(float);
  k_attention<<<dim3(SEQ / 32, NUM_HEADS), 512, attn_lds, stream>>>(
      Qh, Kh, Vt, rel_b, attnh);

  k_out_gemm<<<dim3(SEQ / 128, D_MODEL / 128), 256, 0, stream>>>(attnh, woT, out);
}
